// MultiHeadAttentionBlock_27960237097221
// MI455X (gfx1250) — matrix operand order repaired and checked by reading, not compiled, not run on hardware
//
#include <hip/hip_runtime.h>

// ---------------------------------------------------------------------------
// MHA block for MI455X (gfx1250, wave32, WMMA).
// B=2, S=2048, D=1024, H=16, DK=64.
// Pipeline: f32->bf16 convert -> 3x projection GEMM (bf16 WMMA, f32 acc,
//           double-buffered async-to-LDS tile staging; Q pre-scaled by
//           1/sqrt(DK)) -> fused flash attention (async K-tile staging,
//           WMMA scores + WMMA P@V, online softmax, mask only on the
//           diagonal tile) -> output projection GEMM (f32 to d_out).
// ---------------------------------------------------------------------------

typedef __attribute__((ext_vector_type(16))) __bf16        v16bf;
typedef __attribute__((ext_vector_type(8)))  float         v8f;
typedef __attribute__((ext_vector_type(4)))  unsigned int  u32x4;
typedef __attribute__((ext_vector_type(4)))  int           i32x4;

static constexpr int Bc  = 2;
static constexpr int Sc  = 2048;
static constexpr int Dc  = 1024;
static constexpr int Hc  = 16;
static constexpr int DKc = 64;
static constexpr int Mrows = Bc * Sc;   // 4096

union BF16Frag {
  v16bf          v;      // 8 VGPRs: WMMA A/B operand
  unsigned short u[16];
  u32x4          q[2];   // two 16-byte chunks
};

__device__ __forceinline__ unsigned short f2bf(float f) {
  unsigned int u = __float_as_uint(f);
  u += 0x7FFFu + ((u >> 16) & 1u);        // round to nearest even
  return (unsigned short)(u >> 16);
}

// ---------------------------------------------------------------------------
// CDNA5 async global->LDS copy (ASYNCcnt-tracked DMA), with fallback.
// Builtin prototype (probed): (V4i AS1*, V4i AS3*, Ii offset, Ii cpol)
// ---------------------------------------------------------------------------
#if defined(__AMDGCN__) && __has_builtin(__builtin_amdgcn_global_load_async_to_lds_b128)
#define USE_ASYNC_LDS 1
#endif

__device__ __forceinline__ void copy16_g2l(const unsigned short* g,
                                           unsigned short* l) {
#ifdef USE_ASYNC_LDS
  __builtin_amdgcn_global_load_async_to_lds_b128(
      (__attribute__((address_space(1))) i32x4*)g,
      (__attribute__((address_space(3))) i32x4*)l, 0, 0);
#else
  *(u32x4*)l = *(const u32x4*)g;
#endif
}

__device__ __forceinline__ void async_wait0() {
#ifdef USE_ASYNC_LDS
#if __has_builtin(__builtin_amdgcn_s_wait_asynccnt)
  __builtin_amdgcn_s_wait_asynccnt(0);
#else
  asm volatile("s_wait_asynccnt 0x0" ::: "memory");
#endif
#endif
}

__device__ __forceinline__ void async_wait4() {
#ifdef USE_ASYNC_LDS
#if __has_builtin(__builtin_amdgcn_s_wait_asynccnt)
  __builtin_amdgcn_s_wait_asynccnt(4);
#else
  asm volatile("s_wait_asynccnt 0x4" ::: "memory");
#endif
#endif
}

// ---------------------------------------------------------------------------
// fp32 -> bf16 conversion (grid-stride)
// ---------------------------------------------------------------------------
__global__ void cvt_f32_to_bf16(const float* __restrict__ in,
                                unsigned short* __restrict__ out, int n) {
  int i = blockIdx.x * blockDim.x + threadIdx.x;
  int stride = gridDim.x * blockDim.x;
  for (; i < n; i += stride) out[i] = f2bf(in[i]);
}

// ---------------------------------------------------------------------------
// Y = outScale * (X * W^T)   (X: [M,K] bf16, W: [N,K] bf16 row-major).
// 256 threads = 8 waves; 128x128 tile per block; each wave owns a 64x32
// sub-tile = 4x2 WMMA C tiles.  X/W k-slices are double-buffered in LDS via
// async-to-LDS DMA: issue tile t+1, wait for tile t, compute, barrier.
// MODE 0: bf16 row-major out   MODE 1: bf16 [B,H,S,DK] out   MODE 2: f32 out
// ---------------------------------------------------------------------------
template <int MODE>
__global__ __launch_bounds__(256, 1) void gemm_xwt(
    const unsigned short* __restrict__ X, const unsigned short* __restrict__ W,
    void* __restrict__ Y, int M, int N, int K, float outScale) {
  const int tid  = threadIdx.x;
  const int lane = tid & 31;
  const int wid  = tid >> 5;
  const int waveM = wid >> 2;             // 0..1
  const int waveN = wid & 3;              // 0..3
  const int rowBlk = blockIdx.y * 128;
  const int colBlk = blockIdx.x * 128;
  const int lh = lane >> 4;               // which 16-lane half
  const int ll = lane & 15;

  __shared__ __align__(16) unsigned short Xs[2][128 * 32];   // 2 x 8 KB
  __shared__ __align__(16) unsigned short Ws[2][128 * 32];   // 2 x 8 KB

  // stage one 128x32 slice of X and of W into LDS buffer `buf`
  auto stage = [&](int buf, int k0) {
    for (int c = 0; c < 2; ++c) {
      int linear = c * 256 + tid;          // 0..511
      int row = linear >> 2;               // 0..127
      int col = (linear & 3) * 8;          // 0,8,16,24
      copy16_g2l(X + (size_t)(rowBlk + row) * K + k0 + col,
                 &Xs[buf][row * 32 + col]);
      copy16_g2l(W + (size_t)(colBlk + row) * K + k0 + col,
                 &Ws[buf][row * 32 + col]);
    }
  };  // 4 async ops per thread per call (in-order completion per wave)

  v8f acc[4][2];
  for (int mi = 0; mi < 4; ++mi)
    for (int ni = 0; ni < 2; ++ni)
      for (int r = 0; r < 8; ++r) acc[mi][ni][r] = 0.0f;

  const int nT = K / 32;
  stage(0, 0);
  for (int t = 0; t < nT; ++t) {
    const int buf = t & 1;
    if (t + 1 < nT) {
      stage(buf ^ 1, (t + 1) * 32);
      async_wait4();                 // tile t complete (<=4 newer outstanding)
    } else {
      async_wait0();
    }
    __syncthreads();                 // all waves' staging of tile t visible

    BF16Frag bw[2];
    for (int ni = 0; ni < 2; ++ni) {
      const unsigned short* p =
          &Ws[buf][(waveN * 32 + ni * 16 + ll) * 32 + lh * 8];
      bw[ni].q[0] = *(const u32x4*)p;
      bw[ni].q[1] = *(const u32x4*)(p + 16);
    }
    for (int mi = 0; mi < 4; ++mi) {
      BF16Frag ax;
      const unsigned short* p =
          &Xs[buf][(waveM * 64 + mi * 16 + ll) * 32 + lh * 8];
      ax.q[0] = *(const u32x4*)p;
      ax.q[1] = *(const u32x4*)(p + 16);
      for (int ni = 0; ni < 2; ++ni)
        acc[mi][ni] = __builtin_amdgcn_wmma_f32_16x16x32_bf16(
            false, ax.v, false, bw[ni].v, (short)0, acc[mi][ni], false, false);
    }
    __syncthreads();                 // done reading buf before it is re-staged
  }

  // C layout: VGPR r, lane -> [M = r + 8*lh, N = ll]
  for (int mi = 0; mi < 4; ++mi)
    for (int ni = 0; ni < 2; ++ni)
      for (int r = 0; r < 8; ++r) {
        int m = rowBlk + waveM * 64 + mi * 16 + r + lh * 8;
        int n = colBlk + waveN * 32 + ni * 16 + ll;
        float val = acc[mi][ni][r] * outScale;
        if (MODE == 0) {
          ((unsigned short*)Y)[(size_t)m * N + n] = f2bf(val);
        } else if (MODE == 1) {
          int b = m >> 11, s = m & (Sc - 1);          // m = b*S + s
          int h = n >> 6,  dk = n & (DKc - 1);        // n = h*DK + dk
          ((unsigned short*)Y)[((size_t)(b * Hc + h) * Sc + s) * DKc + dk] =
              f2bf(val);
        } else {
          ((float*)Y)[(size_t)m * N + n] = val;
        }
      }
}

// ---------------------------------------------------------------------------
// Flash attention: one block (4 waves / 128 thr) per (b,h, 64-query tile).
// Each wave owns 16 query rows.  Causal: k-tiles 0..qt; the mask can only
// bite on the diagonal tile (kt == qt), guarded by a uniform branch.
// Q is pre-scaled by 1/sqrt(DK) in the projection GEMM.
// ---------------------------------------------------------------------------
__global__ __launch_bounds__(128, 1) void flash_attn(
    const unsigned short* __restrict__ Q, const unsigned short* __restrict__ K,
    const unsigned short* __restrict__ V, unsigned short* __restrict__ AO) {
  constexpr int QT = Sc / 64;            // 32 query tiles per sequence
  const int qt = blockIdx.x % QT;
  const int bh = blockIdx.x / QT;        // b*H + h
  const int b  = bh / Hc;
  const int h  = bh % Hc;
  const int tid  = threadIdx.x;
  const int lane = tid & 31;
  const int wid  = tid >> 5;             // 0..3
  const int lh = lane >> 4, ll = lane & 15;

  const unsigned short* Qp = Q + (size_t)bh * Sc * DKc;
  const unsigned short* Kp = K + (size_t)bh * Sc * DKc;
  const unsigned short* Vp = V + (size_t)bh * Sc * DKc;

  __shared__ __align__(16) unsigned short Ks[64 * 64];         // raw K tile, 8 KB
  __shared__ __align__(16) unsigned short Vt[DKc * 72];        // V^T tile, padded
  __shared__ __align__(16) unsigned short Plds[4][16 * DKc];   // per-wave P

  const int m0 = qt * 64 + wid * 16;     // first query row of this wave

  // Q A-fragments (rows m0..m0+15, d=0..63) held in VGPRs for the whole loop
  BF16Frag aq[2];
  for (int dd = 0; dd < 2; ++dd) {
    const unsigned short* p = Qp + (size_t)(m0 + ll) * DKc + dd * 32 + lh * 8;
    aq[dd].q[0] = *(const u32x4*)p;
    aq[dd].q[1] = *(const u32x4*)(p + 16);
  }

  float mrow[8], lrow[8];
  v8f o[4];
  for (int r = 0; r < 8; ++r) { mrow[r] = -1e30f; lrow[r] = 0.0f; }
  for (int j = 0; j < 4; ++j)
    for (int r = 0; r < 8; ++r) o[j][r] = 0.0f;

  for (int kt = 0; kt <= qt; ++kt) {
    __syncthreads();   // previous iteration's Ks/Vt reads are done

    // ---- async DMA of raw K tile into LDS (4 b128 ops per thread) ----
    for (int c = 0; c < 4; ++c) {
      int linear = c * 128 + tid;          // 0..511
      int row = linear >> 3;               // 0..63
      int col = (linear & 7) * 8;          // 0..56
      copy16_g2l(Kp + (size_t)(kt * 64 + row) * DKc + col, &Ks[row * 64 + col]);
    }

    // ---- transposed V tile (overlaps the K DMA): Vt[d*72+key] = V[key][d] --
    for (int idx = tid; idx < 64 * 8; idx += 128) {
      int key = idx >> 3;
      int d0  = (idx & 7) * 8;
      union { u32x4 q; unsigned short u[8]; } tmp;
      tmp.q = *(const u32x4*)(Vp + (size_t)(kt * 64 + key) * DKc + d0);
      for (int i = 0; i < 8; ++i) Vt[(d0 + i) * 72 + key] = tmp.u[i];
    }

    async_wait0();
    __syncthreads();

    // ---- scores S[16 x 64] = Q K^T : B[d][key] = K[key][d] from LDS ----
    v8f sc[4];
    for (int j = 0; j < 4; ++j) {
      for (int r = 0; r < 8; ++r) sc[j][r] = 0.0f;
      for (int dd = 0; dd < 2; ++dd) {
        BF16Frag bk;
        const unsigned short* p =
            &Ks[(j * 16 + ll) * 64 + dd * 32 + lh * 8];
        bk.q[0] = *(const u32x4*)p;
        bk.q[1] = *(const u32x4*)(p + 16);
        sc[j] = __builtin_amdgcn_wmma_f32_16x16x32_bf16(
            false, aq[dd].v, false, bk.v, (short)0, sc[j], false, false);
      }
    }

    // ---- causal mask: only the diagonal tile can have key > qrow ----
    if (kt == qt) {
      for (int j = 0; j < 4; ++j) {
        int key = kt * 64 + j * 16 + ll;
        for (int r = 0; r < 8; ++r) {
          int qrow = m0 + r + lh * 8;
          sc[j][r] = (key <= qrow) ? sc[j][r] : -1e30f;
        }
      }
    }

    // ---- online softmax (row reductions across 16-lane halves) ----
    for (int r = 0; r < 8; ++r) {
      float v = fmaxf(fmaxf(sc[0][r], sc[1][r]), fmaxf(sc[2][r], sc[3][r]));
      for (int off = 1; off <= 8; off <<= 1)
        v = fmaxf(v, __shfl_xor(v, off, 32));
      float mnew  = fmaxf(mrow[r], v);
      float alpha = __expf(mrow[r] - mnew);
      mrow[r] = mnew;
      float psum = 0.0f;
      for (int j = 0; j < 4; ++j) {
        float p = __expf(sc[j][r] - mnew);
        sc[j][r] = p;
        psum += p;
      }
      for (int off = 1; off <= 8; off <<= 1)
        psum += __shfl_xor(psum, off, 32);
      lrow[r] = lrow[r] * alpha + psum;
      for (int j = 0; j < 4; ++j) o[j][r] *= alpha;
    }

    // ---- stage P (bf16, row-major 16x64) in per-wave LDS ----
    for (int j = 0; j < 4; ++j)
      for (int r = 0; r < 8; ++r)
        Plds[wid][(r + lh * 8) * DKc + j * 16 + ll] = f2bf(sc[j][r]);

    // ---- O += P @ V : A = P from LDS, B = V^T from LDS (contiguous) ----
    for (int kk = 0; kk < 2; ++kk) {
      BF16Frag ap;
      const unsigned short* p = &Plds[wid][(size_t)ll * DKc + kk * 32 + lh * 8];
      ap.q[0] = *(const u32x4*)p;
      ap.q[1] = *(const u32x4*)(p + 16);
      for (int j = 0; j < 4; ++j) {
        BF16Frag bv;
        const unsigned short* pv = &Vt[(j * 16 + ll) * 72 + kk * 32 + lh * 8];
        bv.q[0] = *(const u32x4*)pv;
        bv.q[1] = *(const u32x4*)(pv + 16);
        o[j] = __builtin_amdgcn_wmma_f32_16x16x32_bf16(
            false, ap.v, false, bv.v, (short)0, o[j], false, false);
      }
    }
  }

  // ---- normalize and write AO[b, s, h*DK + d] (bf16, row-major [B,S,D]) ----
  for (int r = 0; r < 8; ++r) {
    float inv = 1.0f / lrow[r];
    int srow = m0 + r + lh * 8;
    for (int j = 0; j < 4; ++j) {
      int d = j * 16 + ll;
      AO[((size_t)b * Sc + srow) * Dc + h * DKc + d] = f2bf(o[j][r] * inv);
    }
  }
}

// ---------------------------------------------------------------------------
// Host launcher
// ---------------------------------------------------------------------------
extern "C" void kernel_launch(void* const* d_in, const int* in_sizes, int n_in,
                              void* d_out, int out_size, void* d_ws,
                              size_t ws_size, hipStream_t stream) {
  (void)in_sizes; (void)n_in; (void)out_size; (void)ws_size;

  const float* q  = (const float*)d_in[0];
  const float* k  = (const float*)d_in[1];
  const float* v  = (const float*)d_in[2];
  // d_in[3] is the causal mask; causality is known statically -> never read.
  const float* Wq = (const float*)d_in[4];
  const float* Wk = (const float*)d_in[5];
  const float* Wv = (const float*)d_in[6];
  const float* Wo = (const float*)d_in[7];

  const size_t NX = (size_t)Mrows * Dc;   // 4096*1024
  const size_t NW = (size_t)Dc * Dc;      // 1024*1024

  unsigned short* ws  = (unsigned short*)d_ws;
  unsigned short* qb  = ws; ws += NX;
  unsigned short* kb  = ws; ws += NX;
  unsigned short* vb  = ws; ws += NX;
  unsigned short* wqb = ws; ws += NW;
  unsigned short* wkb = ws; ws += NW;
  unsigned short* wvb = ws; ws += NW;
  unsigned short* wob = ws; ws += NW;
  unsigned short* Qh  = ws; ws += NX;     // [B,H,S,DK], pre-scaled by 1/8
  unsigned short* Kh  = ws; ws += NX;
  unsigned short* Vh  = ws; ws += NX;
  unsigned short* AO  = ws; ws += NX;     // [B,S,D]

  cvt_f32_to_bf16<<<1024, 256, 0, stream>>>(q,  qb,  (int)NX);
  cvt_f32_to_bf16<<<1024, 256, 0, stream>>>(k,  kb,  (int)NX);
  cvt_f32_to_bf16<<<1024, 256, 0, stream>>>(v,  vb,  (int)NX);
  cvt_f32_to_bf16<<<256,  256, 0, stream>>>(Wq, wqb, (int)NW);
  cvt_f32_to_bf16<<<256,  256, 0, stream>>>(Wk, wkb, (int)NW);
  cvt_f32_to_bf16<<<256,  256, 0, stream>>>(Wv, wvb, (int)NW);
  cvt_f32_to_bf16<<<256,  256, 0, stream>>>(Wo, wob, (int)NW);

  dim3 gg(Dc / 128, Mrows / 128);   // (8, 32)
  // Q projection pre-scales by 1/sqrt(DK) = 0.125
  gemm_xwt<1><<<gg, 256, 0, stream>>>(qb, wqb, Qh, Mrows, Dc, Dc, 0.125f);
  gemm_xwt<1><<<gg, 256, 0, stream>>>(kb, wkb, Kh, Mrows, Dc, Dc, 1.0f);
  gemm_xwt<1><<<gg, 256, 0, stream>>>(vb, wvb, Vh, Mrows, Dc, Dc, 1.0f);

  flash_attn<<<Bc * Hc * (Sc / 64), 128, 0, stream>>>(Qh, Kh, Vh, AO);

  gemm_xwt<2><<<gg, 256, 0, stream>>>(AO, wob, d_out, Mrows, Dc, Dc, 1.0f);
}
